// JukeboxAttention_4337916969040
// MI455X (gfx1250) — matrix operand order repaired and checked by reading, not compiled, not run on hardware
//
#include <hip/hip_runtime.h>

typedef __attribute__((ext_vector_type(16))) __bf16 bf16x16;
typedef __attribute__((ext_vector_type(8)))  __bf16 bf16x8;
typedef __attribute__((ext_vector_type(8)))  float  f32x8;
typedef __attribute__((ext_vector_type(4)))  float  f32x4;
typedef __attribute__((ext_vector_type(8)))  unsigned short u16x8;
typedef __attribute__((ext_vector_type(4)))  unsigned int   u32x4;
typedef __attribute__((ext_vector_type(8)))  int            i32x8;
typedef __attribute__((ext_vector_type(4)))  int            i32x4;

#define B_SZ   2
#define L_CTX  8192
#define EMB    2048
#define HEADS  16
#define HD     128
#define BLOCKS 16
#define BC     512
#define QKV_LD (3 * 2048)   // 6144

// ---------------------------------------------------------------- helpers
static __device__ __forceinline__ unsigned short f2bf(float f) {
  union { float f; unsigned int u; } c; c.f = f;
  unsigned int u = c.u;
  unsigned int r = u + 0x7fffu + ((u >> 16) & 1u);  // round-to-nearest-even
  return (unsigned short)(r >> 16);
}

static __device__ __forceinline__ bf16x16 cat8(bf16x8 lo, bf16x8 hi) {
  bf16x16 r;
#pragma unroll
  for (int i = 0; i < 8; ++i) { r[i] = lo[i]; r[i + 8] = hi[i]; }
  return r;
}

static __device__ __forceinline__ f32x8 fzero8() {
  f32x8 v;
#pragma unroll
  for (int i = 0; i < 8; ++i) v[i] = 0.0f;
  return v;
}

static __device__ __forceinline__ bf16x8 ld_bf8(const unsigned short* p) {
  return *(const bf16x8*)p;
}

static __device__ __forceinline__ f32x8 wmma_bf16(bf16x16 a, bf16x16 b, f32x8 c) {
  // D = A(16x32) * B(32x16) + C, fp32 accumulate
  return __builtin_amdgcn_wmma_f32_16x16x32_bf16(false, a, false, b, (short)0, c,
                                                 false, false);
}

// ---- Tensor Data Mover: 2D tile load (bf16), row-padded into LDS ----------
// Packs the D# descriptor per CDNA5 ISA:
//   group0: count=1 | lds_addr | global_addr(57b) | type=2
//   group1: data_size=2B, pad_enable (4 DWORDs pad every 16 DWORDs = 64B row
//           -> LDS row stride 80B = 40 bf16), tile 32 x rows, tensor stride.
// clang-23 toolchain: 6-arg builtin (g0, g1, g2, g3, g4, cpol).
static __device__ __forceinline__ void tdm_load_2d_bf16(
    const unsigned short* gaddr, unsigned lds_off, unsigned tile_rows,
    unsigned row_stride_elems) {
  unsigned long long ga = (unsigned long long)(size_t)gaddr;
  u32x4 g0;
  g0[0] = 1u;                                   // count=1 (valid), user mode
  g0[1] = lds_off;                              // LDS byte address
  g0[2] = (unsigned)ga;                         // global addr [31:0]
  g0[3] = (unsigned)((ga >> 32) & 0x01FFFFFFu)  // global addr [56:32]
          | 0x80000000u;                        // type=2 ("image")
  const unsigned dim0 = 1u << 20, dim1 = 1u << 20;  // generous (no OOB clip)
  i32x8 g1;
  g1[0] = (int)((1u << 16)      // data_size = 1 -> 2 bytes
                | (1u << 20)    // pad_enable
                | (3u << 22)    // pad_interval: every 16 DWORDs (one 64B row)
                | (3u << 25));  // pad_amount: 4 DWORDs (16B)
  g1[1] = (int)((dim0 & 0xFFFFu) << 16);                         // tensor_dim0 lo
  g1[2] = (int)(((dim0 >> 16) & 0xFFFFu) | ((dim1 & 0xFFFFu) << 16));
  g1[3] = (int)(((dim1 >> 16) & 0xFFFFu) | (32u << 16));         // tile_dim0=32
  g1[4] = (int)tile_rows;                                        // tile_dim1
  g1[5] = (int)row_stride_elems;                                 // dim0_stride lo
  g1[6] = 0;
  g1[7] = 0;
  i32x4 z4 = {0, 0, 0, 0};
  i32x8 z8 = {0, 0, 0, 0, 0, 0, 0, 0};
  __builtin_amdgcn_tensor_load_to_lds(g0, g1, z4, z4, z8, 0);
}

// ---------------------------------------------------------------- fp32->bf16
__global__ __launch_bounds__(256) void cvt_f32_bf16(
    const float* __restrict__ in, unsigned short* __restrict__ out, int n) {
  int i = (blockIdx.x * 256 + threadIdx.x) * 8;
  if (i + 8 <= n) {
    f32x4 f0 = *(const f32x4*)(in + i);
    f32x4 f1 = *(const f32x4*)(in + i + 4);
    u16x8 v;
#pragma unroll
    for (int j = 0; j < 4; ++j) { v[j] = f2bf(f0[j]); v[j + 4] = f2bf(f1[j]); }
    *(u16x8*)(out + i) = v;
  }
}

// ---------------------------------------------------------------- GEMM
// C[M,N] = A[M,K](bf16) @ W[K,N](fp32) + bias[N]
// A tiles: TDM DMA global->LDS (double buffered, padded stride 40).
// W tiles: staged transposed (W^T) with on-the-fly fp32->bf16 conversion.
template <bool OUT_BF16>
__global__ __launch_bounds__(256) void gemm_bf16_wmma(
    const unsigned short* __restrict__ A, const float* __restrict__ W,
    const float* __restrict__ bias, void* __restrict__ Cp, int M, int N, int K) {
  constexpr int AS = 40;  // padded LDS row stride (bf16 elems)
  __shared__ __align__(16) unsigned short Abuf[2][128 * AS];
  __shared__ __align__(16) unsigned short Bbuf[2][128 * AS];

  const int m0 = blockIdx.y * 128;
  const int n0 = blockIdx.x * 128;
  const int t = threadIdx.x;
  const int w = t >> 5, lane = t & 31;
  const int lh = lane >> 4, lm = lane & 15;
  const int wm = (w & 3) * 32;   // 4 waves along M
  const int wn = (w >> 2) * 64;  // 2 waves along N

  f32x8 acc[2][4];
#pragma unroll
  for (int i = 0; i < 2; ++i)
#pragma unroll
    for (int j = 0; j < 4; ++j) acc[i][j] = fzero8();

  const unsigned short* Abase = A + (size_t)m0 * K;

  // prologue: DMA first A tile into buffer 0
  if (w == 0)
    tdm_load_2d_bf16(Abase, (unsigned)(size_t)&Abuf[0][0], 128, (unsigned)K);

  int p = 0;
  for (int k0 = 0; k0 < K; k0 += 32, p ^= 1) {
    // ---- stage W^T tile: Bbuf[p][n][k], n in [0,128), k in [0,32)
#pragma unroll
    for (int r = 0; r < 2; ++r) {
      int s = t + r * 256;
      int kk = s >> 4;
      int n8 = (s & 15) * 8;
      const float* g = W + (size_t)(k0 + kk) * N + n0 + n8;
      f32x4 f0 = *(const f32x4*)g;
      f32x4 f1 = *(const f32x4*)(g + 4);
#pragma unroll
      for (int i = 0; i < 4; ++i) {
        Bbuf[p][(n8 + i) * AS + kk]     = f2bf(f0[i]);
        Bbuf[p][(n8 + 4 + i) * AS + kk] = f2bf(f1[i]);
      }
      if (k0 + 32 < K) __builtin_prefetch(g + (size_t)32 * N, 0, 1);
    }
    // A tile (issued last iteration) must have landed before the barrier
    if (w == 0) __builtin_amdgcn_s_wait_tensorcnt(0);
    __syncthreads();
    // DMA next A tile into the other buffer while we compute this one
    if (w == 0 && k0 + 32 < K)
      tdm_load_2d_bf16(Abase + k0 + 32, (unsigned)(size_t)&Abuf[p ^ 1][0], 128,
                       (unsigned)K);

    // ---- compute: 8 WMMAs per wave per K-step
    bf16x16 afrag[2];
#pragma unroll
    for (int i = 0; i < 2; ++i) {
      // A frag: lane row = M, elems = K {lh*8..+7} and {16+lh*8..+7}
      const unsigned short* q = &Abuf[p][(wm + i * 16 + lm) * AS + lh * 8];
      afrag[i] = cat8(ld_bf8(q), ld_bf8(q + 16));
    }
#pragma unroll
    for (int j = 0; j < 4; ++j) {
      // B frag: lane col = N, elems = K {lh*8..+7} and {16+lh*8..+7}
      const unsigned short* q = &Bbuf[p][(wn + j * 16 + lm) * AS + lh * 8];
      bf16x16 bfrag = cat8(ld_bf8(q), ld_bf8(q + 16));
#pragma unroll
      for (int i = 0; i < 2; ++i) acc[i][j] = wmma_bf16(afrag[i], bfrag, acc[i][j]);
    }
  }

  // ---- epilogue: bias + store (C frag: lane col n, rows e + 8*lh)
#pragma unroll
  for (int j = 0; j < 4; ++j) {
    int n = n0 + wn + j * 16 + lm;
    float bv = bias[n];
#pragma unroll
    for (int i = 0; i < 2; ++i) {
#pragma unroll
      for (int e = 0; e < 8; ++e) {
        int row = m0 + wm + i * 16 + e + 8 * lh;
        float val = acc[i][j][e] + bv;
        if (OUT_BF16)
          ((unsigned short*)Cp)[(size_t)row * N + n] = f2bf(val);
        else
          ((float*)Cp)[(size_t)row * N + n] = val;
      }
    }
  }
}

// ---------------------------------------------------------------- attention
// One WG per (batch, block, head, 128-query tile). 8 waves x 16 q-rows.
// Flash-style online softmax over key chunks of 128 within the 512 block.
__global__ __launch_bounds__(256) void block_attn(
    const unsigned short* __restrict__ qkv, unsigned short* __restrict__ ctx) {
  const int wg = blockIdx.x;
  const int qt = wg & 3;
  const int head = (wg >> 2) & 15;
  const int blk = (wg >> 6) & 15;
  const int b = wg >> 10;

  const int t = threadIdx.x;
  const int w = t >> 5, lane = t & 31;
  const int lh = lane >> 4, lm = lane & 15;
  const int q0 = qt * 128 + w * 16;  // wave's first query row within block

  const size_t tokbase = (size_t)b * L_CTX + (size_t)blk * BC;
  const int qcol = head * HD;
  const int kcol = EMB + head * HD;
  const int vcol = 2 * EMB + head * HD;

  __shared__ __align__(16) unsigned short VT[128 * 136];    // V^T: [d][key]
  __shared__ __align__(16) unsigned short PB[8 * 16 * 40];  // per-wave P tile

  // ---- load this wave's Q fragments (16 rows x 128 d) once, from global
  bf16x16 qf[4];
#pragma unroll
  for (int dt = 0; dt < 4; ++dt) {
    const unsigned short* g =
        qkv + (tokbase + q0 + lm) * (size_t)QKV_LD + qcol + dt * 32 + lh * 8;
    qf[dt] = cat8(ld_bf8(g), ld_bf8(g + 16));
  }

  f32x8 accO[8];
#pragma unroll
  for (int i = 0; i < 8; ++i) accO[i] = fzero8();
  float mrow[8], lrow[8];
#pragma unroll
  for (int e = 0; e < 8; ++e) { mrow[e] = -3.0e38f; lrow[e] = 0.0f; }

  const float s2 = 0.08838834764831845f;  // HD^-0.5 (= SCALE^2 on logits)
  const int kc_end = qt * 128 + 128;      // causal: chunks through own q tile

  for (int kc0 = 0; kc0 < kc_end; kc0 += 128) {
    __syncthreads();
    // ---- cooperative stage of V^T chunk [128 d][128 keys]
#pragma unroll
    for (int r = 0; r < 8; ++r) {
      int s = t + r * 256;
      int key = s >> 4;
      int d0 = (s & 15) * 8;
      const unsigned short* g =
          qkv + (tokbase + kc0 + key) * (size_t)QKV_LD + vcol + d0;
      u16x8 v = *(const u16x8*)g;
#pragma unroll
      for (int i = 0; i < 8; ++i) VT[(d0 + i) * 136 + key] = v[i];
    }
    __syncthreads();

    if (kc0 <= q0 + 15) {  // wave needs this key chunk (causal skip)
      // ---- S = Q K^T : 16 q x 128 keys (K frags straight from global)
      f32x8 accS[8];
#pragma unroll
      for (int kt = 0; kt < 8; ++kt) accS[kt] = fzero8();
#pragma unroll
      for (int kt = 0; kt < 8; ++kt) {
#pragma unroll
        for (int dt = 0; dt < 4; ++dt) {
          const unsigned short* g = qkv +
              (tokbase + kc0 + kt * 16 + lm) * (size_t)QKV_LD + kcol +
              dt * 32 + lh * 8;
          bf16x16 bfrag = cat8(ld_bf8(g), ld_bf8(g + 16));
          accS[kt] = wmma_bf16(qf[dt], bfrag, accS[kt]);
        }
      }
      // ---- scale + exact reference mask (-1e9)
#pragma unroll
      for (int kt = 0; kt < 8; ++kt) {
        int kj = kc0 + kt * 16 + lm;
#pragma unroll
        for (int e = 0; e < 8; ++e) {
          int qi = q0 + e + 8 * lh;
          float v = accS[kt][e] * s2;
          accS[kt][e] = (kj > qi) ? -1.0e9f : v;
        }
      }
      // ---- row max (per-16-lane-half butterflies match C layout)
      float mloc[8];
#pragma unroll
      for (int e = 0; e < 8; ++e) {
        float v = accS[0][e];
#pragma unroll
        for (int kt = 1; kt < 8; ++kt) v = fmaxf(v, accS[kt][e]);
        mloc[e] = v;
      }
#pragma unroll
      for (int msk = 1; msk <= 8; msk <<= 1)
#pragma unroll
        for (int e = 0; e < 8; ++e)
          mloc[e] = fmaxf(mloc[e], __shfl_xor(mloc[e], msk, 32));

      float scl[8];
#pragma unroll
      for (int e = 0; e < 8; ++e) {
        float mnew = fmaxf(mrow[e], mloc[e]);
        scl[e] = __expf(mrow[e] - mnew);
        mrow[e] = mnew;
      }
      // ---- p = exp(S - m), row sums
      float rsum[8];
#pragma unroll
      for (int e = 0; e < 8; ++e) rsum[e] = 0.0f;
#pragma unroll
      for (int kt = 0; kt < 8; ++kt)
#pragma unroll
        for (int e = 0; e < 8; ++e) {
          float pp = __expf(accS[kt][e] - mrow[e]);
          accS[kt][e] = pp;
          rsum[e] += pp;
        }
#pragma unroll
      for (int msk = 1; msk <= 8; msk <<= 1)
#pragma unroll
        for (int e = 0; e < 8; ++e) rsum[e] += __shfl_xor(rsum[e], msk, 32);
#pragma unroll
      for (int e = 0; e < 8; ++e) lrow[e] = lrow[e] * scl[e] + rsum[e];
#pragma unroll
      for (int i = 0; i < 8; ++i)
#pragma unroll
        for (int e = 0; e < 8; ++e) accO[i][e] *= scl[e];

      // ---- O += P V : C-layout -> A-layout via tiny per-wave LDS bounce
      unsigned short* pb = &PB[w * 16 * 40];
#pragma unroll
      for (int s = 0; s < 4; ++s) {
#pragma unroll
        for (int half = 0; half < 2; ++half) {
          int kt = 2 * s + half;
#pragma unroll
          for (int e = 0; e < 8; ++e)
            pb[(e + 8 * lh) * 40 + half * 16 + lm] = f2bf(accS[kt][e]);
        }
        const unsigned short* pa = pb + lm * 40 + lh * 8;
        bf16x16 afrag = cat8(ld_bf8(pa), ld_bf8(pa + 16));
#pragma unroll
        for (int dtile = 0; dtile < 8; ++dtile) {
          const unsigned short* vb =
              &VT[(dtile * 16 + lm) * 136 + s * 32 + lh * 8];
          bf16x16 bfrag = cat8(ld_bf8(vb), ld_bf8(vb + 16));
          accO[dtile] = wmma_bf16(afrag, bfrag, accO[dtile]);
        }
      }
    }
  }

  // ---- normalize + store ctx (bf16, merged-head layout [tok][E])
#pragma unroll
  for (int e = 0; e < 8; ++e) lrow[e] = 1.0f / lrow[e];
#pragma unroll
  for (int dtile = 0; dtile < 8; ++dtile) {
#pragma unroll
    for (int e = 0; e < 8; ++e) {
      int row = q0 + e + 8 * lh;
      ctx[(tokbase + row) * (size_t)EMB + head * HD + dtile * 16 + lm] =
          f2bf(accO[dtile][e] * lrow[e]);
    }
  }
}

// ---------------------------------------------------------------- launch
extern "C" void kernel_launch(void* const* d_in, const int* in_sizes, int n_in,
                              void* d_out, int out_size, void* d_ws,
                              size_t ws_size, hipStream_t stream) {
  (void)in_sizes; (void)n_in; (void)out_size; (void)ws_size;
  const float* x      = (const float*)d_in[0];
  const float* attn_w = (const float*)d_in[1];
  const float* attn_b = (const float*)d_in[2];
  const float* proj_w = (const float*)d_in[3];
  const float* proj_b = (const float*)d_in[4];

  const int M = B_SZ * L_CTX;  // 16384

  unsigned short* xb  = (unsigned short*)d_ws;                  // [16384,2048]
  unsigned short* qkv = xb + (size_t)M * EMB;                   // [16384,6144]
  unsigned short* ctx = qkv + (size_t)M * QKV_LD;               // [16384,2048]
  float* out = (float*)d_out;

  // 0) x -> bf16 (one cheap pass; makes both GEMM A-paths pure TDM copies)
  cvt_f32_bf16<<<dim3((M * EMB) / (256 * 8)), 256, 0, stream>>>(x, xb, M * EMB);

  // 1) qkv = x @ c_attn_w + c_attn_b  (bf16 out)
  gemm_bf16_wmma<true>
      <<<dim3(3 * EMB / 128, M / 128), 256, 0, stream>>>(xb, attn_w, attn_b,
                                                         qkv, M, 3 * EMB, EMB);

  // 2) block-causal flash attention -> ctx (bf16)
  block_attn<<<dim3(B_SZ * BLOCKS * HEADS * (BC / 128)), 256, 0, stream>>>(qkv,
                                                                           ctx);

  // 3) out = ctx @ c_proj_w + c_proj_b (fp32 out)
  gemm_bf16_wmma<false>
      <<<dim3(EMB / 128, M / 128), 256, 0, stream>>>(ctx, proj_w, proj_b, out,
                                                     M, EMB, EMB);
}
